// SpatialAttention_14491219657087
// MI455X (gfx1250) — matrix operand order repaired and checked by reading, not compiled, not run on hardware
//
#include <hip/hip_runtime.h>

// ---- problem constants (match reference) ----
#define N_NODES 2048
#define T_STEPS 4
#define D_HEAD  64
#define H_HEADS 4
#define DMODEL  (H_HEADS * D_HEAD)       // 256
#define SROW    2052                     // padded LDS row stride (floats); multiple of 4 keeps 16B alignment

typedef __attribute__((ext_vector_type(16))) _Float16 v16h;
typedef __attribute__((ext_vector_type(8)))  _Float16 v8h;
typedef __attribute__((ext_vector_type(8)))  float    v8f;
typedef __attribute__((ext_vector_type(4)))  float    v4f;

// ---------------- helpers ----------------
static __device__ __forceinline__ float wave_max(float v) {
#pragma unroll
    for (int off = 16; off >= 1; off >>= 1)
        v = fmaxf(v, __shfl_xor(v, off, 32));
    return v;
}
static __device__ __forceinline__ float wave_sum(float v) {
#pragma unroll
    for (int off = 16; off >= 1; off >>= 1)
        v += __shfl_xor(v, off, 32);
    return v;
}
static __device__ __forceinline__ v16h cat8(v8h lo, v8h hi) {
    v16h r;
#pragma unroll
    for (int i = 0; i < 8; ++i) { r[i] = lo[i]; r[i + 8] = hi[i]; }
    return r;
}
// B-fragment: two 8-f16 runs (K=8h..8h+7 and 16+8h..16+8h+7 for lane half h)
static __device__ __forceinline__ v16h load_bfrag(const _Float16* p) {
    return cat8(*(const v8h*)p, *(const v8h*)(p + 16));
}

// row max / sum-of-exp for 16 rows of S; b128 LDS reads (4 floats/lane/op)
static __device__ __forceinline__ void row_stats(const float* S, float* st_max, float* st_sum, int lane) {
#pragma unroll 1
    for (int r = 0; r < 16; ++r) {
        const float* row = S + r * SROW;
        float mx = -3.0e38f;
#pragma unroll 4
        for (int c = lane * 4; c < N_NODES; c += 128) {
            const v4f v = *(const v4f*)(row + c);
            mx = fmaxf(fmaxf(fmaxf(v[0], v[1]), fmaxf(v[2], v[3])), mx);
        }
        mx = wave_max(mx);
        float sm = 0.0f;
#pragma unroll 4
        for (int c = lane * 4; c < N_NODES; c += 128) {
            const v4f v = *(const v4f*)(row + c);
            sm += __expf(v[0] - mx) + __expf(v[1] - mx) + __expf(v[2] - mx) + __expf(v[3] - mx);
        }
        sm = wave_sum(sm);
        if (lane == 0) { st_max[r] = mx; st_sum[r] = sm; }
    }
    __syncthreads();
}

// P = exp(S - max), O += P @ V  (4 x 16x16 tiles over D=64), scale by 1/sum, store.
static __device__ __forceinline__ void pv_pass(const float* S, const float* st_max, const float* st_sum,
                                               const _Float16* __restrict__ VT, float* __restrict__ dst,
                                               int t, int h, int i0, int lane)
{
    const int mrow = lane & 15;
    const int hi   = lane >> 4;   // lane half
    const int kbA  = hi * 8;      // A-fragment K base (ISA 16-bit A layout)
    const int kbB  = hi * 8;      // B-fragment K base
    const float pmax = st_max[mrow];
    const float* Srow = S + mrow * SROW;

    v8f o0 = {}, o1 = {}, o2 = {}, o3 = {};
    for (int jj = 0; jj < N_NODES / 32; ++jj) {
        const int c0 = jj * 32;
        // A-fragment of P: 4 x b128 LDS reads, exp on the fly
        const v4f s0 = *(const v4f*)(Srow + c0 + kbA);
        const v4f s1 = *(const v4f*)(Srow + c0 + kbA + 4);
        const v4f s2 = *(const v4f*)(Srow + c0 + 16 + kbA);
        const v4f s3 = *(const v4f*)(Srow + c0 + 16 + kbA + 4);
        v16h pa;
#pragma unroll
        for (int i = 0; i < 4; ++i) {
            pa[i]      = (_Float16)__expf(s0[i] - pmax);
            pa[i + 4]  = (_Float16)__expf(s1[i] - pmax);
            pa[i + 8]  = (_Float16)__expf(s2[i] - pmax);
            pa[i + 12] = (_Float16)__expf(s3[i] - pmax);
        }
        o0 = __builtin_amdgcn_wmma_f32_16x16x32_f16(false, pa, false,
                 load_bfrag(VT + (size_t)(0 * 16 + mrow) * N_NODES + c0 + kbB), (short)0, o0, false, false);
        o1 = __builtin_amdgcn_wmma_f32_16x16x32_f16(false, pa, false,
                 load_bfrag(VT + (size_t)(1 * 16 + mrow) * N_NODES + c0 + kbB), (short)0, o1, false, false);
        o2 = __builtin_amdgcn_wmma_f32_16x16x32_f16(false, pa, false,
                 load_bfrag(VT + (size_t)(2 * 16 + mrow) * N_NODES + c0 + kbB), (short)0, o2, false, false);
        o3 = __builtin_amdgcn_wmma_f32_16x16x32_f16(false, pa, false,
                 load_bfrag(VT + (size_t)(3 * 16 + mrow) * N_NODES + c0 + kbB), (short)0, o3, false, false);
    }

    // out layout (from reference permute/reshape): flat = h*N*T*D + n*T*D + t*D + d
#pragma unroll
    for (int r = 0; r < 8; ++r) {
        const int m = r + hi * 8;
        const float inv = 1.0f / st_sum[m];
        size_t base = (size_t)h * ((size_t)N_NODES * T_STEPS * D_HEAD)
                    + (size_t)(i0 + m) * (T_STEPS * D_HEAD)
                    + (size_t)t * D_HEAD + mrow;
        dst[base + 0]  = o0[r] * inv;
        dst[base + 16] = o1[r] * inv;
        dst[base + 32] = o2[r] * inv;
        dst[base + 48] = o3[r] * inv;
    }
}

// ---------------- kernel 1: QKV projection ----------------
// q,k stored (t,h,n,d) f16; v stored transposed (t,h,d,n) f16 so P@V B-fragments are contiguous.
__global__ void __launch_bounds__(DMODEL)
proj_qkv(const float* __restrict__ x,
         const float* __restrict__ Wq, const float* __restrict__ bq,
         const float* __restrict__ Wk, const float* __restrict__ bk,
         const float* __restrict__ Wv, const float* __restrict__ bv,
         _Float16* __restrict__ q, _Float16* __restrict__ k, _Float16* __restrict__ vt)
{
    __shared__ float xs[D_HEAD];
    const int bx = blockIdx.x;            // n*T + t
    const int n = bx >> 2, t = bx & 3;
    const int c = threadIdx.x;            // output channel 0..255
    if (c < D_HEAD) xs[c] = x[(size_t)bx * D_HEAD + c];
    __syncthreads();

    const int h = c >> 6, d = c & 63;
    float aq = bq[c], ak = bk[c], av = bv[c];
    const float* wq = Wq + (size_t)c * D_HEAD;
    const float* wk = Wk + (size_t)c * D_HEAD;
    const float* wv = Wv + (size_t)c * D_HEAD;
#pragma unroll 8
    for (int i = 0; i < D_HEAD; ++i) {
        const float xv = xs[i];
        aq = fmaf(xv, wq[i], aq);
        ak = fmaf(xv, wk[i], ak);
        av = fmaf(xv, wv[i], av);
    }
    const size_t th = (size_t)(t * H_HEADS + h);
    const size_t qi = (th * N_NODES + n) * D_HEAD + d;
    q[qi] = (_Float16)aq;
    k[qi] = (_Float16)ak;
    vt[(th * D_HEAD + d) * N_NODES + n] = (_Float16)av;
}

// ---------------- kernel 2: fused dual-softmax attention ----------------
// One wave32 per (t, h, 16-row block). S kept in LDS (16 x 2052 f32).
__global__ void __launch_bounds__(32)
attn_fused(const _Float16* __restrict__ qg, const _Float16* __restrict__ kg,
           const _Float16* __restrict__ vtg,
           const float* __restrict__ attw, const unsigned char* __restrict__ attm,
           float* __restrict__ out, float* __restrict__ out_raw, float* __restrict__ att_slice)
{
    extern __shared__ float smem[];
    float* S      = smem;                   // 16 x SROW
    float* st_max = smem + 16 * SROW;       // 16
    float* st_sum = st_max + 16;            // 16

    const int t = blockIdx.z, h = blockIdx.y;
    const int i0 = blockIdx.x * 16;
    const int lane = threadIdx.x;
    const int mrow = lane & 15;
    const int hi   = lane >> 4;
    const int kbA  = hi * 8;
    const int kbB  = hi * 8;

    const size_t th = (size_t)(t * H_HEADS + h);
    const _Float16* Q  = qg  + th * N_NODES * D_HEAD;
    const _Float16* K  = kg  + th * N_NODES * D_HEAD;
    const _Float16* VT = vtg + th * (size_t)D_HEAD * N_NODES;

    // Q A-fragments for both K-steps (K = 0..31, 32..63), row = i0 + mrow
    v16h a0, a1;
    {
        const _Float16* qr = Q + (size_t)(i0 + mrow) * D_HEAD;
        a0 = cat8(*(const v8h*)(qr + kbA),      *(const v8h*)(qr + 16 + kbA));
        a1 = cat8(*(const v8h*)(qr + 32 + kbA), *(const v8h*)(qr + 48 + kbA));
    }

    // scores: S[m][j*16 + n] = (Q . K^T) / 8
    for (int j = 0; j < N_NODES / 16; ++j) {
        const _Float16* kr = K + (size_t)(j * 16 + mrow) * D_HEAD;  // B-fragment: N = mrow
        v16h b0 = load_bfrag(kr + kbB);
        v16h b1 = load_bfrag(kr + 32 + kbB);
        v8f c = {};
        c = __builtin_amdgcn_wmma_f32_16x16x32_f16(false, a0, false, b0, (short)0, c, false, false);
        c = __builtin_amdgcn_wmma_f32_16x16x32_f16(false, a1, false, b1, (short)0, c, false, false);
#pragma unroll
        for (int r = 0; r < 8; ++r)
            S[(r + hi * 8) * SROW + j * 16 + mrow] = c[r] * 0.125f;
    }
    __syncthreads();

    // ---- pass A: raw softmax ----
    row_stats(S, st_max, st_sum, lane);

    // raw attention slice output for (t = T-1, h = 0), b128 in/out
    if (t == T_STEPS - 1 && h == 0) {
#pragma unroll 1
        for (int r = 0; r < 16; ++r) {
            const float mx = st_max[r], inv = 1.0f / st_sum[r];
            const size_t base = (size_t)(i0 + r) * N_NODES;
#pragma unroll 2
            for (int c = lane * 4; c < N_NODES; c += 128) {
                const v4f s = *(const v4f*)(S + r * SROW + c);
                v4f o;
#pragma unroll
                for (int i = 0; i < 4; ++i) o[i] = __expf(s[i] - mx) * inv;
                *(v4f*)(att_slice + base + c) = o;
            }
        }
    }

    pv_pass(S, st_max, st_sum, VT, out_raw, t, h, i0, lane);
    __syncthreads();

    // ---- transform in place: S <- mask ? -inf : S * w ----
    // Single streaming read of the 256MB+64MB tensors: float4 weights, uchar4 mask.
    const float*         wr = attw + (th * N_NODES + i0) * (size_t)N_NODES;
    const unsigned char* mr = attm + (th * N_NODES + i0) * (size_t)N_NODES;
#pragma unroll 1
    for (int r = 0; r < 16; ++r) {
        float* srow = S + r * SROW;
        const float*        wrow = wr + (size_t)r * N_NODES;
        const unsigned int* mrow = (const unsigned int*)(mr + (size_t)r * N_NODES);
#pragma unroll 2
        for (int c4 = lane; c4 < N_NODES / 4; c4 += 32) {
            const v4f s  = *(const v4f*)(srow + c4 * 4);
            const v4f w4 = *(const v4f*)(wrow + c4 * 4);
            const unsigned int m4 = mrow[c4];
            v4f o;
            o[0] = (m4 & 0x000000ffu) ? -__builtin_inff() : s[0] * w4[0];
            o[1] = (m4 & 0x0000ff00u) ? -__builtin_inff() : s[1] * w4[1];
            o[2] = (m4 & 0x00ff0000u) ? -__builtin_inff() : s[2] * w4[2];
            o[3] = (m4 & 0xff000000u) ? -__builtin_inff() : s[3] * w4[3];
            *(v4f*)(srow + c4 * 4) = o;
        }
    }
    __syncthreads();

    // ---- pass B: weighted+masked softmax ----
    row_stats(S, st_max, st_sum, lane);
    pv_pass(S, st_max, st_sum, VT, out, t, h, i0, lane);
}

// ---------------- launcher ----------------
extern "C" void kernel_launch(void* const* d_in, const int* in_sizes, int n_in,
                              void* d_out, int out_size, void* d_ws, size_t ws_size,
                              hipStream_t stream) {
    (void)in_sizes; (void)n_in; (void)out_size; (void)ws_size;

    const float* x    = (const float*)d_in[0];
    const float* Wq   = (const float*)d_in[1];
    const float* bq   = (const float*)d_in[2];
    const float* Wk   = (const float*)d_in[3];
    const float* bk   = (const float*)d_in[4];
    const float* Wv   = (const float*)d_in[5];
    const float* bv   = (const float*)d_in[6];
    const float* attw = (const float*)d_in[7];
    const unsigned char* attm = (const unsigned char*)d_in[8];  // jax bool -> 1 byte

    const size_t QKV = (size_t)T_STEPS * H_HEADS * N_NODES * D_HEAD;  // 2,097,152
    _Float16* q  = (_Float16*)d_ws;
    _Float16* k  = q + QKV;
    _Float16* vt = k + QKV;            // ~12.6 MB total workspace

    float* out       = (float*)d_out;          // [0, QKV)
    float* out_raw   = out + QKV;              // [QKV, 2*QKV)
    float* att_slice = out_raw + QKV;          // [2*QKV, 2*QKV + N*N)

    proj_qkv<<<dim3(N_NODES * T_STEPS), dim3(DMODEL), 0, stream>>>(
        x, Wq, bq, Wk, bk, Wv, bv, q, k, vt);

    const size_t smem_bytes = (size_t)(16 * SROW + 32) * sizeof(float);
    attn_fused<<<dim3(N_NODES / 16, H_HEADS, T_STEPS), dim3(32), smem_bytes, stream>>>(
        q, k, vt, attw, attm, out, out_raw, att_slice);
}
